// PointConv_87540023427829
// MI455X (gfx1250) — matrix operand order repaired and checked by reading, not compiled, not run on hardware
//
#include <hip/hip_runtime.h>

typedef __attribute__((ext_vector_type(2)))  float    v2f;
typedef __attribute__((ext_vector_type(4)))  float    v4f;
typedef __attribute__((ext_vector_type(8)))  float    v8f;
typedef __attribute__((ext_vector_type(16))) _Float16 v16h;

#define WAVES_PER_BLOCK 8
#define NODES_PER_GROUP 16
constexpr int NPTS = 50000;
constexpr int KNB  = 32;      // neighbors per node (deg == 32 everywhere)
constexpr int CMID = 64;
constexpr int COUT = 64;

__device__ __forceinline__ float celu1(float x) {
    return x > 0.0f ? x : (__expf(x) - 1.0f);   // jax.nn.celu, alpha=1
}

__device__ __forceinline__ float fix_out(float y) {
    // nan_to_num(posinf=1e4, neginf=-1e4)
    if (y != y)                 return 0.0f;
    if (y ==  __builtin_inff()) return  10000.0f;
    if (y == -__builtin_inff()) return -10000.0f;
    return y;
}

__global__ __launch_bounds__(256) void
pointconv_wmma_kernel(const float* __restrict__ x_in,
                      const float* __restrict__ pos_in,
                      const float* __restrict__ W1,   // [3,16]
                      const float* __restrict__ W2,   // [16,64]
                      const float* __restrict__ W3,   // [64,64]
                      const float* __restrict__ b3,   // [64]
                      const int*   __restrict__ in_index,
                      float*       __restrict__ out)  // [N,64]
{
    __shared__ float shH[WAVES_PER_BLOCK][256];    // 16x16 hidden transpose scratch
    __shared__ float shP[WAVES_PER_BLOCK][1024];   // 16x64 P rows

    const int  tid  = threadIdx.x;
    const int  wave = tid >> 5;
    const int  lane = tid & 31;
    const int  l16  = lane & 15;
    const bool hiH  = lane >= 16;

    const int group   = blockIdx.x * WAVES_PER_BLOCK + wave;
    const int ngroups = NPTS / NODES_PER_GROUP;    // 3125, exact
    if (group >= ngroups) return;                  // wave-uniform: EXEC stays all-1s

    // ---------------- stage weights (unconditional loads + select) ----------------
    // Layer-1 B: W1 (3x16) zero-padded to 4x16, f32 16x16x4 B layout.
    v2f b1;
    {
        const float w0 = W1[l16];            // K0 row
        const float w1 = W1[16 + l16];       // K1 row
        const float w2 = W1[32 + l16];       // K2 row
        b1.x = hiH ? w2 : w0;                // lanes0-15: K0 | lanes16-31: K2
        b1.y = hiH ? 0.0f : w1;              // lanes0-15: K1 | lanes16-31: K3(pad)
    }
    // Layer-2 B tiles: W2 (16x64) padded to 32x64, f16 16x16x32 B, contiguous K.
    // lanes 0-15 carry K=0..15 (real rows); lanes 16-31 carry K=16..31 (zero pad).
    v16h bw2_0, bw2_1, bw2_2, bw2_3;
    {
        const _Float16 zh = (_Float16)0.0f;
        #pragma unroll
        for (int j = 0; j < 16; ++j) {
            const float a = W2[j * CMID +  0 + l16];   // plain saddr+offset loads
            const float b = W2[j * CMID + 16 + l16];
            const float c = W2[j * CMID + 32 + l16];
            const float d = W2[j * CMID + 48 + l16];
            bw2_0[j] = hiH ? zh : (_Float16)a;
            bw2_1[j] = hiH ? zh : (_Float16)b;
            bw2_2[j] = hiH ? zh : (_Float16)c;
            bw2_3[j] = hiH ? zh : (_Float16)d;
        }
    }
    const float b3v0 = b3[ 0 + l16];
    const float b3v1 = b3[16 + l16];
    const float b3v2 = b3[32 + l16];
    const float b3v3 = b3[48 + l16];

    // ---------------- per-node edge MLP + weighted aggregation ----------------
    #pragma unroll 1
    for (int ng = 0; ng < NODES_PER_GROUP; ++ng) {
        const int node = group * NODES_PER_GROUP + ng;
        const int eidx = node * KNB + lane;
        const int src  = in_index[eidx];
        __builtin_prefetch(&in_index[eidx + KNB], 0, 1);  // next node's indices

        const float pox = pos_in[node * 3 + 0];           // wave-uniform -> s_load
        const float poy = pos_in[node * 3 + 1];
        const float poz = pos_in[node * 3 + 2];
        float px = pos_in[src * 3 + 0] - pox;
        float py = pos_in[src * 3 + 1] - poy;
        float pz = pos_in[src * 3 + 2] - poz;
        px = (px != px) ? 0.0f : px;                      // nan_to_num
        py = (py != py) ? 0.0f : py;
        pz = (pz != pz) ? 0.0f : pz;
        const float xj = x_in[src] * (1.0f / KNB);        // deg == 32

        float acc0 = 0.f, acc1 = 0.f, acc2 = 0.f, acc3 = 0.f;

        #pragma unroll
        for (int t = 0; t < 2; ++t) {                     // two 16-edge tiles
            // ---- A1 (pos_local tile, f32 16x4 A layout) via shuffles ----
            const int srcl = t * 16 + l16;
            const float sx = __shfl(px, srcl, 32);
            const float sy = __shfl(py, srcl, 32);
            const float sz = __shfl(pz, srcl, 32);
            v2f a1;
            a1.x = hiH ? sz : sx;                         // K2 : K0
            a1.y = hiH ? 0.0f : sy;                       // K3(pad) : K1
            v8f z8 = {0.f,0.f,0.f,0.f,0.f,0.f,0.f,0.f};
            v8f h  = __builtin_amdgcn_wmma_f32_16x16x4_f32(
                         false, a1, false, b1, (short)0, z8, false, false);

            // ---- CELU, then D-layout -> A-layout transpose through LDS ----
            #pragma unroll
            for (int v = 0; v < 8; ++v) {
                const int row = v + (hiH ? 8 : 0);        // edge within tile
                shH[wave][row * 16 + l16] = celu1(h[v]);
            }
            asm volatile("s_wait_dscnt 0x0" ::: "memory");

            v16h a2;
            // f16 A 16x32, contiguous K: lanes 0-15 K=0..15, lanes 16-31 K pad
            #pragma unroll
            for (int j = 0; j < 16; ++j)
                a2[j] = hiH ? (_Float16)0.0f : (_Float16)shH[wave][l16 * 16 + j];
            asm volatile("" ::: "memory");

            // per-edge scalars for the 8 rows this half-wave reduces
            float xs0 = __shfl(xj, t*16 + 0 + (hiH ? 8 : 0), 32);
            float xs1 = __shfl(xj, t*16 + 1 + (hiH ? 8 : 0), 32);
            float xs2 = __shfl(xj, t*16 + 2 + (hiH ? 8 : 0), 32);
            float xs3 = __shfl(xj, t*16 + 3 + (hiH ? 8 : 0), 32);
            float xs4 = __shfl(xj, t*16 + 4 + (hiH ? 8 : 0), 32);
            float xs5 = __shfl(xj, t*16 + 5 + (hiH ? 8 : 0), 32);
            float xs6 = __shfl(xj, t*16 + 6 + (hiH ? 8 : 0), 32);
            float xs7 = __shfl(xj, t*16 + 7 + (hiH ? 8 : 0), 32);

            // ---- layer 2: 16->64 via f16 WMMA + CELU + weighted row-sum ----
            auto l2 = [&](const v16h& bw) -> float {
                v8f zc = {0.f,0.f,0.f,0.f,0.f,0.f,0.f,0.f};
                v8f m  = __builtin_amdgcn_wmma_f32_16x16x32_f16(
                             false, a2, false, bw, (short)0, zc, false, false);
                return xs0*celu1(m[0]) + xs1*celu1(m[1]) + xs2*celu1(m[2]) +
                       xs3*celu1(m[3]) + xs4*celu1(m[4]) + xs5*celu1(m[5]) +
                       xs6*celu1(m[6]) + xs7*celu1(m[7]);
            };
            acc0 += l2(bw2_0);
            acc1 += l2(bw2_1);
            acc2 += l2(bw2_2);
            acc3 += l2(bw2_3);
        }
        // combine half-wave partials; park P row in LDS
        acc0 += __shfl_xor(acc0, 16, 32);
        acc1 += __shfl_xor(acc1, 16, 32);
        acc2 += __shfl_xor(acc2, 16, 32);
        acc3 += __shfl_xor(acc3, 16, 32);
        if (!hiH) {
            shP[wave][ng * COUT +  0 + l16] = acc0;
            shP[wave][ng * COUT + 16 + l16] = acc1;
            shP[wave][ng * COUT + 32 + l16] = acc2;
            shP[wave][ng * COUT + 48 + l16] = acc3;
        }
    }

    // ---------------- final GEMM: out[16x64] = P[16x64] @ W3[64x64] + b3 ----------------
    asm volatile("s_wait_dscnt 0x0" ::: "memory");
    v8f oacc0, oacc1, oacc2, oacc3;
    #pragma unroll
    for (int v = 0; v < 8; ++v) {
        oacc0[v] = b3v0; oacc1[v] = b3v1; oacc2[v] = b3v2; oacc3[v] = b3v3;
    }

    const int koff = hiH ? 2 : 0;
    #pragma unroll 4
    for (int kc = 0; kc < 16; ++kc) {                  // K=64 in chunks of 4
        const int k0 = kc * 4 + koff;
        v2f a;                                         // A: P rows (row = node)
        a.x = shP[wave][l16 * COUT + k0];
        a.y = shP[wave][l16 * COUT + k0 + 1];
        auto gstep = [&](int n0, v8f c) -> v8f {
            v2f b;                                     // B: W3 in f32 B layout
            b.x = W3[k0 * COUT + n0 + l16];
            b.y = W3[(k0 + 1) * COUT + n0 + l16];
            return __builtin_amdgcn_wmma_f32_16x16x4_f32(
                       false, a, false, b, (short)0, c, false, false);
        };
        oacc0 = gstep( 0, oacc0);
        oacc1 = gstep(16, oacc1);
        oacc2 = gstep(32, oacc2);
        oacc3 = gstep(48, oacc3);
    }

    // ---- nan_to_num, transpose through LDS (P-buffer dead), coalesced b128 stores ----
    asm volatile("s_wait_dscnt 0x0" ::: "memory");
    #pragma unroll
    for (int v = 0; v < 8; ++v) {
        const int row = v + (hiH ? 8 : 0);             // node within group
        shP[wave][row * COUT +  0 + l16] = fix_out(oacc0[v]);
        shP[wave][row * COUT + 16 + l16] = fix_out(oacc1[v]);
        shP[wave][row * COUT + 32 + l16] = fix_out(oacc2[v]);
        shP[wave][row * COUT + 48 + l16] = fix_out(oacc3[v]);
    }
    asm volatile("s_wait_dscnt 0x0" ::: "memory");
    float* obase = &out[group * NODES_PER_GROUP * COUT];   // contiguous 1024 floats
    #pragma unroll
    for (int i = 0; i < 8; ++i) {
        const int idx = i * 128 + lane * 4;
        v4f vvec;
        vvec.x = shP[wave][idx + 0];
        vvec.y = shP[wave][idx + 1];
        vvec.z = shP[wave][idx + 2];
        vvec.w = shP[wave][idx + 3];
        *(v4f*)&obase[idx] = vvec;                     // global_store_b128, coalesced
    }
}

extern "C" void kernel_launch(void* const* d_in, const int* in_sizes, int n_in,
                              void* d_out, int out_size, void* d_ws, size_t ws_size,
                              hipStream_t stream) {
    const float* x_in     = (const float*)d_in[0];
    const float* pos_in   = (const float*)d_in[1];
    const float* W1       = (const float*)d_in[2];
    const float* W2       = (const float*)d_in[3];
    const float* W3       = (const float*)d_in[4];
    const float* b3       = (const float*)d_in[5];
    // d_in[6] = batch_in (unused: single batch), d_in[8] = out_index (implicit)
    const int*   in_index = (const int*)d_in[7];
    float*       out      = (float*)d_out;

    const int ngroups = NPTS / NODES_PER_GROUP;                 // 3125
    const int blocks  = (ngroups + WAVES_PER_BLOCK - 1) / WAVES_PER_BLOCK;  // 391
    hipLaunchKernelGGL(pointconv_wmma_kernel, dim3(blocks), dim3(256), 0, stream,
                       x_in, pos_in, W1, W2, W3, b3, in_index, out);
}
